// BDH_58368605553068
// MI455X (gfx1250) — matrix operand order repaired and checked by reading, not compiled, not run on hardware
//
#include <hip/hip_runtime.h>

typedef __bf16 bf16;
typedef __attribute__((ext_vector_type(16))) __bf16 v16bf;
typedef __attribute__((ext_vector_type(8)))  float  v8f;

constexpr int T  = 1024;
constexpr int D  = 192;
constexpr int NH = 4;
constexpr int NN = 3072;
constexpr int V  = 256;
constexpr float EPS = 1e-5f;

__device__ __forceinline__ bf16  f2bf(float f) { return (bf16)f; }
__device__ __forceinline__ float bf2f(bf16 h)  { return (float)h; }

#define WMMA_BF16(a, b, c) \
  __builtin_amdgcn_wmma_f32_16x16x32_bf16(false, (a), false, (b), (short)0, (c), false, false)

// ---------------------------------------------------------------------------
// Fragment loaders (wave32). K order here is contiguous, not the ISA 7.12.2 table order.
// A (16x32 bf16): lane L holds row M=L&15, K = k0 + 16*(L>>4) .. +15
// (same K order as B) -> one contiguous 32B chunk per lane.
__device__ __forceinline__ v16bf load_A_frag(const bf16* __restrict__ base, int ld,
                                             int tm, int k0, int lane) {
  int row  = tm + (lane & 15);
  int koff = k0 + ((lane >> 4) << 4);
  const bf16* p = base + (size_t)row * ld + koff;
  v16bf a;
#pragma unroll
  for (int i = 0; i < 16; ++i) a[i] = p[i];
  return a;
}

// B (32x16 bf16): lane L holds column N = L&15, K = k0 + 16*(L>>4) .. +15.
// All B operands are stored so that columns of B are contiguous rows of `base`
// (i.e. base = Bᵀ row-major) -> one contiguous 32B chunk per lane.
__device__ __forceinline__ v16bf load_B_rows(const bf16* __restrict__ base, int ld,
                                             int tn, int k0, int lane) {
  int row = tn + (lane & 15);
  const bf16* p = base + (size_t)row * ld + k0 + ((lane >> 4) << 4);
  v16bf b;
#pragma unroll
  for (int i = 0; i < 16; ++i) b[i] = p[i];
  return b;
}

// One K-step of a 32x64 (2x4-tile) block: 6 frag loads, 8 WMMAs.
__device__ __forceinline__ void mma_step_2x4(const bf16* __restrict__ A, int lda, int tm,
                                             const bf16* __restrict__ Bt, int ldb, int tn,
                                             int k0, int lane, v8f acc[2][4]) {
  v16bf a0 = load_A_frag(A, lda, tm, k0, lane);
  v16bf a1 = load_A_frag(A, lda, tm + 16, k0, lane);
#pragma unroll
  for (int j = 0; j < 4; ++j) {
    v16bf b = load_B_rows(Bt, ldb, tn + 16 * j, k0, lane);
    acc[0][j] = WMMA_BF16(a0, b, acc[0][j]);
    acc[1][j] = WMMA_BF16(a1, b, acc[1][j]);
  }
}

// C/D f32 16x16: col = lane&15, row = 8*(lane>>4) + vgpr.
// ---------------------------------------------------------------------------

// Transpose-convert: in (batch x R x C, f32) -> out (batch x C x R, bf16).
__global__ void k_cvt_t(const float* __restrict__ in, bf16* __restrict__ out,
                        int R, int C) {
  int b = blockIdx.y;
  int i = blockIdx.x * blockDim.x + threadIdx.x;
  if (i >= R * C) return;
  int r = i / C, c = i - r * C;
  out[(size_t)b * R * C + (size_t)c * R + r] = f2bf(in[(size_t)b * R * C + i]);
}

__global__ void k_embed_add(const int* __restrict__ idx, const float* __restrict__ emb,
                            const float* __restrict__ pos, float* __restrict__ out) {
  int i = blockIdx.x * blockDim.x + threadIdx.x;  // < T*D
  int t = i / D, d = i - t * D;
  out[i] = emb[(size_t)idx[t] * D + d] + pos[i];
}

// LayerNorm over rows of length D=192. blockDim = 64, 3 elems/thread.
// Optional residual add; optional f32 / bf16 / transposed-bf16 ([D][T]) outputs.
__global__ void k_ln_rows(const float* __restrict__ in, const float* __restrict__ resid,
                          float* __restrict__ outf, bf16* __restrict__ outb,
                          bf16* __restrict__ outbT) {
  __shared__ float sh1[64], sh2[64];
  int row = blockIdx.x, tid = threadIdx.x;
  const float* p = in + (size_t)row * D;
  const float* r = resid ? resid + (size_t)row * D : nullptr;
  float v[3], s1 = 0.f, s2 = 0.f;
#pragma unroll
  for (int i = 0; i < 3; ++i) {
    float x = p[tid + 64 * i];
    if (r) x += r[tid + 64 * i];
    v[i] = x; s1 += x; s2 += x * x;
  }
  sh1[tid] = s1; sh2[tid] = s2;
  __syncthreads();
  s1 = 0.f; s2 = 0.f;
  for (int i = 0; i < 64; ++i) { s1 += sh1[i]; s2 += sh2[i]; }
  float m = s1 / D;
  float inv = rsqrtf(s2 / D - m * m + EPS);
#pragma unroll
  for (int i = 0; i < 3; ++i) {
    int d = tid + 64 * i;
    float y = (v[i] - m) * inv;
    if (outf)  outf[(size_t)row * D + d] = y;
    if (outb)  outb[(size_t)row * D + d] = f2bf(y);
    if (outbT) outbT[(size_t)d * T + row] = f2bf(y);  // only used when rows == T
  }
}

// x_sparse[h][t][n] = relu(sum_d x[t][d] * decx[h][d][n]).  M=T,K=D,N=NN.
__global__ void k_xsparse(const bf16* __restrict__ xbf, const bf16* __restrict__ wxT,
                          bf16* __restrict__ xsp) {
  int lane = threadIdx.x;
  int tm = blockIdx.x * 32, tn = blockIdx.y * 64, h = blockIdx.z;
  const bf16* Bt = wxT + (size_t)h * D * NN;  // [NN][D]
  v8f acc[2][4] = {};
#pragma unroll
  for (int k0 = 0; k0 < D; k0 += 32)
    mma_step_2x4(xbf, D, tm, Bt, D, tn, k0, lane, acc);
  bf16* out = xsp + (size_t)h * T * NN;
  int rb = ((lane >> 4) << 3), cb = (lane & 15);
#pragma unroll
  for (int i = 0; i < 2; ++i)
#pragma unroll
    for (int j = 0; j < 4; ++j)
#pragma unroll
      for (int rr = 0; rr < 8; ++rr) {
        float vv = acc[i][j][rr];
        out[(size_t)(tm + 16 * i + rb + rr) * NN + tn + 16 * j + cb] =
            f2bf(vv > 0.f ? vv : 0.f);
      }
}

// RoPE with in-kernel freqs: f(n) = 2^(-16*(n&~1)/3072) / (2*pi); phase = frac(t*f)*2*pi.
__global__ void k_rope(const bf16* __restrict__ xsp, bf16* __restrict__ qr) {
  int i = blockIdx.x * blockDim.x + threadIdx.x;  // < NH*T*(NN/2)
  const int half = NN / 2;
  int h = i / (T * half);
  int rem = i - h * (T * half);
  int t = rem / half;
  int n0 = (rem - t * half) * 2;
  float f  = exp2f(-16.0f * (float)n0 / (float)NN) * 0.15915494309189535f;
  float ph = (float)t * f;
  ph = (ph - floorf(ph)) * 6.283185307179586f;
  float c = cosf(ph), s = sinf(ph);
  size_t o = ((size_t)h * T + t) * NN + n0;
  float x0 = bf2f(xsp[o]), x1 = bf2f(xsp[o + 1]);
  qr[o]     = f2bf(x0 * c - x1 * s);
  qr[o + 1] = f2bf(x1 * c + x0 * s);
}

// scores[h][t][s] = (sum_n qr[t][n]*qr[s][n]) * (s < t).  M=T,N=T,K=NN.
__global__ void k_scores(const bf16* __restrict__ qr, bf16* __restrict__ sc) {
  int lane = threadIdx.x;
  int tm = blockIdx.x * 32, tn = blockIdx.y * 64, h = blockIdx.z;
  bf16* out = sc + (size_t)h * T * T;
  int rb = ((lane >> 4) << 3), cb = (lane & 15);
  if (tn >= tm + 32) {  // strictly above diagonal block: fully masked
#pragma unroll
    for (int i = 0; i < 2; ++i)
#pragma unroll
      for (int j = 0; j < 4; ++j)
#pragma unroll
        for (int rr = 0; rr < 8; ++rr)
          out[(size_t)(tm + 16 * i + rb + rr) * T + tn + 16 * j + cb] = f2bf(0.f);
    return;
  }
  const bf16* Q = qr + (size_t)h * T * NN;
  v8f acc[2][4] = {};
  for (int k0 = 0; k0 < NN; k0 += 32) {
    __builtin_prefetch(Q + (size_t)(tm + (lane & 15)) * NN + k0 + 32, 0, 1);
    mma_step_2x4(Q, NN, tm, Q, NN, tn, k0, lane, acc);
  }
#pragma unroll
  for (int i = 0; i < 2; ++i)
#pragma unroll
    for (int j = 0; j < 4; ++j)
#pragma unroll
      for (int rr = 0; rr < 8; ++rr) {
        int row = tm + 16 * i + rb + rr, col = tn + 16 * j + cb;
        out[(size_t)row * T + col] = f2bf(col < row ? acc[i][j][rr] : 0.f);
      }
}

// ykv[h][t][d] = sum_s scores[h][t][s] * x[s][d].  M=T,K=T,N=D (causal K-clamp).
__global__ void k_ykv(const bf16* __restrict__ sc, const bf16* __restrict__ xbfT,
                      float* __restrict__ ykv) {
  int lane = threadIdx.x;
  int tm = blockIdx.x * 32, tn = blockIdx.y * 64, h = blockIdx.z;
  const bf16* S = sc + (size_t)h * T * T;
  int kmax = tm + 32;             // rows <= tm+31 need s <= tm+30
  if (kmax > T) kmax = T;
  v8f acc[2][4] = {};
  for (int k0 = 0; k0 < kmax; k0 += 32)
    mma_step_2x4(S, T, tm, xbfT, T, tn, k0, lane, acc);  // Bt = xᵀ [D][T]
  float* out = ykv + (size_t)h * T * D;
  int rb = ((lane >> 4) << 3), cb = (lane & 15);
#pragma unroll
  for (int i = 0; i < 2; ++i)
#pragma unroll
    for (int j = 0; j < 4; ++j)
#pragma unroll
      for (int rr = 0; rr < 8; ++rr)
        out[(size_t)(tm + 16 * i + rb + rr) * D + tn + 16 * j + cb] = acc[i][j][rr];
}

// xy[h][t][n] = relu(sum_d ykv_ln[h][t][d]*decy[h][d][n]) * x_sparse[h][t][n].
__global__ void k_ysxy(const bf16* __restrict__ ykvb, const bf16* __restrict__ wyT,
                       const bf16* __restrict__ xsp, bf16* __restrict__ xy) {
  int lane = threadIdx.x;
  int tm = blockIdx.x * 32, tn = blockIdx.y * 64, h = blockIdx.z;
  const bf16* A  = ykvb + (size_t)h * T * D;
  const bf16* Bt = wyT + (size_t)h * D * NN;  // [NN][D]
  v8f acc[2][4] = {};
#pragma unroll
  for (int k0 = 0; k0 < D; k0 += 32)
    mma_step_2x4(A, D, tm, Bt, D, tn, k0, lane, acc);
  const bf16* XS = xsp + (size_t)h * T * NN;
  bf16* out = xy + (size_t)h * T * NN;
  int rb = ((lane >> 4) << 3), cb = (lane & 15);
#pragma unroll
  for (int i = 0; i < 2; ++i)
#pragma unroll
    for (int j = 0; j < 4; ++j)
#pragma unroll
      for (int rr = 0; rr < 8; ++rr) {
        float ys = acc[i][j][rr];
        ys = ys > 0.f ? ys : 0.f;
        size_t o = (size_t)(tm + 16 * i + rb + rr) * NN + tn + 16 * j + cb;
        out[o] = f2bf(ys * bf2f(XS[o]));
      }
}

// ymlp[t][d] = sum_h sum_n xy[h][t][n] * enc[h][n][d].  M=T,K=NH*NN,N=D.
__global__ void k_ymlp(const bf16* __restrict__ xy, const bf16* __restrict__ encT,
                       float* __restrict__ ymlp) {
  int lane = threadIdx.x;
  int tm = blockIdx.x * 32, tn = blockIdx.y * 64;
  v8f acc[2][4] = {};
  for (int h = 0; h < NH; ++h) {
    const bf16* A  = xy + (size_t)h * T * NN;
    const bf16* Bt = encT + (size_t)h * NN * D;  // [D][NN]
    for (int k0 = 0; k0 < NN; k0 += 32) {
      __builtin_prefetch(A + (size_t)(tm + (lane & 15)) * NN + k0 + 32, 0, 1);
      mma_step_2x4(A, NN, tm, Bt, NN, tn, k0, lane, acc);
    }
  }
  int rb = ((lane >> 4) << 3), cb = (lane & 15);
#pragma unroll
  for (int i = 0; i < 2; ++i)
#pragma unroll
    for (int j = 0; j < 4; ++j)
#pragma unroll
      for (int rr = 0; rr < 8; ++rr)
        ymlp[(size_t)(tm + 16 * i + rb + rr) * D + tn + 16 * j + cb] = acc[i][j][rr];
}

// logits[t][v] = sum_d x[t][d]*lm_head[d][v].  M=T,K=D,N=V -> f32 d_out.
__global__ void k_logits(const bf16* __restrict__ xbf, const bf16* __restrict__ lmhT,
                         float* __restrict__ out) {
  int lane = threadIdx.x;
  int tm = blockIdx.x * 32, tn = blockIdx.y * 64;
  v8f acc[2][4] = {};
#pragma unroll
  for (int k0 = 0; k0 < D; k0 += 32)
    mma_step_2x4(xbf, D, tm, lmhT, D, tn, k0, lane, acc);  // lmhT [V][D]
  int rb = ((lane >> 4) << 3), cb = (lane & 15);
#pragma unroll
  for (int i = 0; i < 2; ++i)
#pragma unroll
    for (int j = 0; j < 4; ++j)
#pragma unroll
      for (int rr = 0; rr < 8; ++rr)
        out[(size_t)(tm + 16 * i + rb + rr) * V + tn + 16 * j + cb] = acc[i][j][rr];
}

extern "C" void kernel_launch(void* const* d_in, const int* in_sizes, int n_in,
                              void* d_out, int out_size, void* d_ws, size_t ws_size,
                              hipStream_t stream) {
  (void)in_sizes; (void)n_in; (void)out_size; (void)ws_size;
  const int*   idx  = (const int*)d_in[0];
  const float* decx = (const float*)d_in[1];
  const float* decy = (const float*)d_in[2];
  const float* enc  = (const float*)d_in[3];
  const float* emb  = (const float*)d_in[4];
  const float* pos  = (const float*)d_in[5];
  const float* lmh  = (const float*)d_in[6];
  float* out = (float*)d_out;

  char* w = (char*)d_ws;
  size_t off = 0;
  auto alloc = [&](size_t bytes) -> void* {
    void* p = w + off;
    off = (off + bytes + 255) & ~(size_t)255;
    return p;
  };
  bf16*  decxT = (bf16*)alloc((size_t)NH * D * NN * 2);  // [h][n][d]
  bf16*  decyT = (bf16*)alloc((size_t)NH * D * NN * 2);  // [h][n][d]
  bf16*  encT  = (bf16*)alloc((size_t)NH * NN * D * 2);  // [h][d][n]
  bf16*  lmhT  = (bf16*)alloc((size_t)D * V * 2);        // [v][d]
  float* xf    = (float*)alloc((size_t)T * D * 4);
  bf16*  xbf   = (bf16*)alloc((size_t)T * D * 2);
  bf16*  xbfT  = (bf16*)alloc((size_t)T * D * 2);        // [d][t]
  float* tmpf  = (float*)alloc((size_t)T * D * 4);
  bf16*  xspb  = (bf16*)alloc((size_t)NH * T * NN * 2);
  bf16*  qrb   = (bf16*)alloc((size_t)NH * T * NN * 2);
  bf16*  scb   = (bf16*)alloc((size_t)NH * T * T * 2);
  float* ykvf  = (float*)alloc((size_t)NH * T * D * 4);
  bf16*  ykvb  = (bf16*)alloc((size_t)NH * T * D * 2);
  float* ymlpf = (float*)alloc((size_t)T * D * 4);
  bf16*  xyb   = qrb;  // qr is dead after k_scores; reuse for xy

  // Transposed bf16 weight copies (one pass each).
  k_cvt_t<<<dim3((D * NN + 255) / 256, NH), dim3(256), 0, stream>>>(decx, decxT, D, NN);
  k_cvt_t<<<dim3((D * NN + 255) / 256, NH), dim3(256), 0, stream>>>(decy, decyT, D, NN);
  k_cvt_t<<<dim3((NN * D + 255) / 256, NH), dim3(256), 0, stream>>>(enc, encT, NN, D);
  k_cvt_t<<<dim3((D * V + 255) / 256, 1), dim3(256), 0, stream>>>(lmh, lmhT, D, V);

  k_embed_add<<<dim3(T * D / 256), dim3(256), 0, stream>>>(idx, emb, pos, tmpf);
  k_ln_rows<<<dim3(T), dim3(64), 0, stream>>>(tmpf, nullptr, xf, xbf, xbfT);

  for (int layer = 0; layer < 4; ++layer) {
    k_xsparse<<<dim3(T / 32, NN / 64, NH), dim3(32), 0, stream>>>(xbf, decxT, xspb);
    k_rope<<<dim3(NH * T * (NN / 2) / 256), dim3(256), 0, stream>>>(xspb, qrb);
    k_scores<<<dim3(T / 32, T / 64, NH), dim3(32), 0, stream>>>(qrb, scb);
    k_ykv<<<dim3(T / 32, D / 64, NH), dim3(32), 0, stream>>>(scb, xbfT, ykvf);
    k_ln_rows<<<dim3(NH * T), dim3(64), 0, stream>>>(ykvf, nullptr, nullptr, ykvb, nullptr);
    k_ysxy<<<dim3(T / 32, NN / 64, NH), dim3(32), 0, stream>>>(ykvb, decyT, xspb, xyb);
    k_ymlp<<<dim3(T / 32, D / 64), dim3(32), 0, stream>>>(xyb, encT, ymlpf);
    k_ln_rows<<<dim3(T), dim3(64), 0, stream>>>(ymlpf, nullptr, tmpf, nullptr, nullptr);
    // x = ln(x + ln(ymlp)); refresh xbf and xᵀ for next layer
    k_ln_rows<<<dim3(T), dim3(64), 0, stream>>>(tmpf, xf, xf, xbf, xbfT);
  }
  k_logits<<<dim3(T / 32, V / 64), dim3(32), 0, stream>>>(xbf, lmhT, out);
}
